// RationalQuadraticSpline_73495480369667
// MI455X (gfx1250) — matrix operand order repaired and checked by reading, not compiled, not run on hardware
//
#include <hip/hip_runtime.h>
#include <hip/hip_bf16.h>

// ---------------- problem constants ----------------
#define D_       32
#define NF_      64
#define KBINS    16
#define BCONST   3.0f
#define NH_      512
#define NB_      4
#define BATCH_   32768
#define MDK_     0.001f
#define MDER_    0.001f
#define MOFF_    1e-6f
#define DIMOUT_  1504          // (NF-D)*(3K-1) = 32*47

// ---------------- tiling constants -----------------
#define BM     64              // batch rows per workgroup
#define HPAD   520             // LDS h row stride (bf16 elems), 512+8 pad (16B)
#define TPAD   192             // LDS theta row stride (f32 elems), 188 padded
#define FCHUNK 4               // spline features per out-GEMM chunk
#define CCOLS  188             // FCHUNK*47 theta cols per chunk
#define NCHUNK 8               // 8*188 = 1504

typedef __attribute__((ext_vector_type(16))) __bf16 v16bf;
typedef __attribute__((ext_vector_type(8)))  float  v8f;

union BF16Frag { v16bf v; uint4 q[2]; };

__device__ __forceinline__ v8f v8zero() {
  v8f z;
#pragma unroll
  for (int i = 0; i < 8; ++i) z[i] = 0.f;
  return z;
}

// A fragment (16x32 bf16, M x K) from LDS row-major h tile.
// 16-bit A layout: lane 0-15 -> M=lane, K runs [8h,8h+8) in V0..V3 and
// [16+8h, 16+8h+8) in V4..V7, h = lane>>4.
__device__ __forceinline__ v16bf load_a_lds(const __hip_bfloat16* h,
                                            int mbase, int kbase) {
  const int lane = threadIdx.x & 31;
  const int m    = mbase + (lane & 15);
  const int half = lane >> 4;
  const __hip_bfloat16* p = h + m * HPAD + kbase + 8 * half;
  BF16Frag u;
  u.q[0] = *(const uint4*)(p);        // K run 0
  u.q[1] = *(const uint4*)(p + 16);   // K run 1 (+16 elems)
  return u.v;
}

// B fragment (32x16 bf16, K x N) from global transposed weights wT[n][k].
// 16-bit B layout: K runs [8h,8h+8) in V0..V3 and [16+8h,16+8h+8) in V4..V7,
// h = lane>>4 (same as A).  n is the per-lane column.
__device__ __forceinline__ v16bf load_b_glb(const __hip_bfloat16* wT,
                                            int n, int kbase, int krows) {
  const int half = (threadIdx.x >> 4) & 1;
  const __hip_bfloat16* p = wT + (size_t)n * krows + kbase + 8 * half;
  BF16Frag u;
  u.q[0] = *(const uint4*)(p);        // K run 0
  u.q[1] = *(const uint4*)(p + 16);   // K run 1 (+16 elems)
  return u.v;
}

__device__ __forceinline__ v8f wmma_bf16(v16bf a, v16bf b, v8f c) {
  return __builtin_amdgcn_wmma_f32_16x16x32_bf16(
      /*neg_a=*/false, a, /*neg_b=*/false, b,
      /*c_mod=*/(short)0, c, /*reuse_a=*/false, /*reuse_b=*/false);
}

// ---------------- weight convert + transpose: f32[K][N] -> bf16[N][K] -----
__global__ void cvt_transpose(const float* __restrict__ src,
                              __hip_bfloat16* __restrict__ dst,
                              int K, int N) {
  size_t i = (size_t)blockIdx.x * 256 + threadIdx.x;
  if (i >= (size_t)K * N) return;
  int n = (int)(i % N);
  int k = (int)(i / N);
  dst[(size_t)n * K + k] = __float2bfloat16(src[i]);
}

// ---------------- fused network + spline kernel ---------------------------
// One workgroup = 64 batch rows, 256 threads = 8 waves, 2 WGs / WGP.
__global__ __launch_bounds__(256)
void nsf_fused(const float* __restrict__ x,
               const float* __restrict__ b_in,
               const float* __restrict__ b_res,
               const float* __restrict__ b_out,
               const __hip_bfloat16* __restrict__ WinT,
               const __hip_bfloat16* __restrict__ WresT,
               const __hip_bfloat16* __restrict__ WoutT,
               float* __restrict__ out) {
  extern __shared__ char smem[];
  __hip_bfloat16* hbuf = (__hip_bfloat16*)smem;                    // BM x HPAD bf16
  float* thetaS = (float*)(smem + (size_t)BM * HPAD * 2);          // BM x TPAD f32

  const int tid   = threadIdx.x;
  const int lane  = tid & 31;
  const int w     = tid >> 5;           // wave id 0..7
  const int sbase = blockIdx.x * BM;

  // ---- stage cond (bf16) into hbuf; copy cond f32 to output y ----
  for (int j = tid; j < BM * D_; j += 256) {
    int s = j >> 5, c = j & 31;
    float v = x[(size_t)(sbase + s) * NF_ + c];
    hbuf[s * HPAD + c] = __float2bfloat16(v);
    out[(size_t)(sbase + s) * NF_ + c] = v;
  }
  __syncthreads();

  // ---- input layer: h = relu(cond @ W_in + b_in), K = 32 (in-place) ----
  {
    v8f acc[4][4];
#pragma unroll
    for (int mt = 0; mt < 4; ++mt)
#pragma unroll
      for (int nt = 0; nt < 4; ++nt) acc[mt][nt] = v8zero();

    v16bf a[4];
#pragma unroll
    for (int mt = 0; mt < 4; ++mt) a[mt] = load_a_lds(hbuf, mt * 16, 0);
#pragma unroll
    for (int nt = 0; nt < 4; ++nt) {
      int n = w * 64 + nt * 16 + (lane & 15);
      v16bf b = load_b_glb(WinT, n, 0, 32);
#pragma unroll
      for (int mt = 0; mt < 4; ++mt) acc[mt][nt] = wmma_bf16(a[mt], b, acc[mt][nt]);
    }
    __syncthreads();   // cond reads complete before cols 0..31 are overwritten
#pragma unroll
    for (int nt = 0; nt < 4; ++nt) {
      int n = w * 64 + nt * 16 + (lane & 15);
      float bias = b_in[n];
#pragma unroll
      for (int mt = 0; mt < 4; ++mt) {
        int mrow = mt * 16 + (lane >> 4) * 8;
#pragma unroll
        for (int r = 0; r < 8; ++r) {
          float vv = acc[mt][nt][r] + bias;
          hbuf[(mrow + r) * HPAD + n] = __float2bfloat16(fmaxf(vv, 0.f));
        }
      }
    }
  }
  __syncthreads();

  // ---- residual blocks: h = relu(h + h @ W_res[l] + b_res[l]) (in-place) ----
  for (int layer = 0; layer < NB_; ++layer) {
    const __hip_bfloat16* wT = WresT + (size_t)layer * NH_ * NH_;
    const float* br = b_res + layer * NH_;

    // prefetch this wave's weight columns into L2/L0 (global_prefetch_b8)
#pragma unroll
    for (int nt = 0; nt < 4; ++nt) {
      int n = w * 64 + nt * 16 + (lane & 15);
      __builtin_prefetch(wT + (size_t)n * NH_, 0, 3);
    }

    v8f acc[4][4];
#pragma unroll
    for (int mt = 0; mt < 4; ++mt)
#pragma unroll
      for (int nt = 0; nt < 4; ++nt) acc[mt][nt] = v8zero();

    for (int kk = 0; kk < NH_; kk += 32) {
      v16bf a[4];
#pragma unroll
      for (int mt = 0; mt < 4; ++mt) a[mt] = load_a_lds(hbuf, mt * 16, kk);
#pragma unroll
      for (int nt = 0; nt < 4; ++nt) {
        int n = w * 64 + nt * 16 + (lane & 15);
        v16bf b = load_b_glb(wT, n, kk, NH_);
#pragma unroll
        for (int mt = 0; mt < 4; ++mt) acc[mt][nt] = wmma_bf16(a[mt], b, acc[mt][nt]);
      }
    }
    __syncthreads();   // all A-fragment reads done before in-place update
    // epilogue: residual add + bias + relu, element-wise RMW by owning lane
#pragma unroll
    for (int nt = 0; nt < 4; ++nt) {
      int n = w * 64 + nt * 16 + (lane & 15);
      float bias = br[n];
#pragma unroll
      for (int mt = 0; mt < 4; ++mt) {
        int mrow = mt * 16 + (lane >> 4) * 8;
#pragma unroll
        for (int r = 0; r < 8; ++r) {
          int m = mrow + r;
          float hv = __bfloat162float(hbuf[m * HPAD + n]);
          float vv = hv + bias + acc[mt][nt][r];
          hbuf[m * HPAD + n] = __float2bfloat16(fmaxf(vv, 0.f));
        }
      }
    }
    __syncthreads();
  }

  // ---- out-GEMM in 8 feature-chunks, fused with spline ----
  const int s_l = tid >> 2;   // 0..63  local sample
  const int f_l = tid & 3;    // 0..3   feature-in-chunk
  float ldet = 0.f;

  for (int c = 0; c < NCHUNK; ++c) {
    // theta chunk cols [c*188, c*188+188), padded to 12 tiles of 16.
    // 12 N-tiles x 2 M-halves = 24 units, 3 per wave (balanced).
    for (int i = 0; i < 3; ++i) {
      int u    = w * 3 + i;
      int nt   = u >> 1;
      int mb   = (u & 1) * 32;
      int ncol = nt * 16 + (lane & 15);     // col within padded chunk
      int ng   = c * CCOLS + ncol;
      if (ng > DIMOUT_ - 1) ng = DIMOUT_ - 1;   // clamp pad lanes (discarded)
      v8f c0 = v8zero(), c1 = v8zero();
      for (int kk = 0; kk < NH_; kk += 32) {
        v16bf a0 = load_a_lds(hbuf, mb, kk);
        v16bf a1 = load_a_lds(hbuf, mb + 16, kk);
        v16bf b  = load_b_glb(WoutT, ng, kk, NH_);
        c0 = wmma_bf16(a0, b, c0);
        c1 = wmma_bf16(a1, b, c1);
      }
      if (ncol < CCOLS) {
        float bias = b_out[c * CCOLS + ncol];
        int mrow0 = (lane >> 4) * 8;
#pragma unroll
        for (int r = 0; r < 8; ++r) {
          thetaS[(mb + mrow0 + r) * TPAD + ncol]      = c0[r] + bias;
          thetaS[(mb + 16 + mrow0 + r) * TPAD + ncol] = c1[r] + bias;
        }
      }
    }
    __syncthreads();

    // ---- spline for feature f = c*4 + f_l of local sample s_l ----
    {
      const float* tp = &thetaS[s_l * TPAD + f_l * 47];
      const int f  = c * FCHUNK + f_l;
      const int sg = sbase + s_l;
      const float xq = x[(size_t)sg * NF_ + D_ + f];

      float kw[17], kh[17], der[17];
      // widths: tp[0..15]
      {
        float mx = tp[0];
#pragma unroll
        for (int i = 1; i < 16; ++i) mx = fmaxf(mx, tp[i]);
        float sum = 0.f;
#pragma unroll
        for (int i = 0; i < 16; ++i) sum += __expf(tp[i] - mx);
        float inv = (1.0f - MDK_ * KBINS) / sum;
        float cum = 0.f;
        kw[0] = -BCONST;
#pragma unroll
        for (int i = 0; i < 16; ++i) {
          cum += __expf(tp[i] - mx) * inv + MDK_;
          kw[i + 1] = 2.f * BCONST * cum - BCONST;
        }
        kw[16] = BCONST;
      }
      // heights: tp[16..31]
      {
        float mx = tp[16];
#pragma unroll
        for (int i = 1; i < 16; ++i) mx = fmaxf(mx, tp[16 + i]);
        float sum = 0.f;
#pragma unroll
        for (int i = 0; i < 16; ++i) sum += __expf(tp[16 + i] - mx);
        float inv = (1.0f - MDK_ * KBINS) / sum;
        float cum = 0.f;
        kh[0] = -BCONST;
#pragma unroll
        for (int i = 0; i < 16; ++i) {
          cum += __expf(tp[16 + i] - mx) * inv + MDK_;
          kh[i + 1] = 2.f * BCONST * cum - BCONST;
        }
        kh[16] = BCONST;
      }
      // derivatives: tp[32..46]
      der[0] = 1.f; der[16] = 1.f;
#pragma unroll
      for (int i = 0; i < 15; ++i) {
        float td = tp[32 + i];
        float sp = (td > 20.f) ? td : __logf(1.f + __expf(td));
        der[i + 1] = MDER_ + sp;
      }

      // bin search (branchless, unrolled)
      int cnt = 0;
#pragma unroll
      for (int i = 0; i < 17; ++i) cnt += (kw[i] <= xq) ? 1 : 0;
      int ia = cnt < 1 ? 1 : (cnt > 16 ? 16 : cnt);
      int ib = ia - 1;
      float xk = 0.f, xk1 = 0.f, yk = 0.f, yk1 = 0.f, dk = 0.f, dk1 = 0.f;
#pragma unroll
      for (int i = 0; i < 17; ++i) {
        bool bb = (i == ib), ba = (i == ia);
        xk  = bb ? kw[i]  : xk;   yk  = bb ? kh[i]  : yk;   dk  = bb ? der[i] : dk;
        xk1 = ba ? kw[i]  : xk1;  yk1 = ba ? kh[i]  : yk1;  dk1 = ba ? der[i] : dk1;
      }

      float wd  = xk1 - xk + MOFF_;
      float xi  = (xq - xk) / wd;
      xi = fminf(fmaxf(xi, 0.f), 1.f);
      float sl  = (yk1 - yk) / wd;
      float xi1 = xi * (1.f - xi);
      float num_g = (yk1 - yk) * (sl * xi * xi + dk * xi1);
      float den_g = sl + (dk1 + dk - 2.f * sl) * xi1 + MOFF_;
      float g = yk + num_g / den_g;
      float omxi = 1.f - xi;
      float num_der = sl * sl * (dk1 * xi * xi + 2.f * sl * xi1 + dk * omxi * omxi);
      float der_g = num_der / (den_g * den_g);

      out[(size_t)sg * NF_ + D_ + f] = g;
      ldet += __logf(der_g + MOFF_);
    }
    __syncthreads();   // protect thetaS before next chunk's GEMM overwrites
  }

  // ---- reduce log-det over the 4 feature lanes of each sample ----
  ldet += __shfl_xor(ldet, 1, 4);
  ldet += __shfl_xor(ldet, 2, 4);
  if (f_l == 0)
    out[(size_t)BATCH_ * NF_ + (sbase + s_l)] = ldet;
}

// ---------------- host launch ---------------------------------------------
extern "C" void kernel_launch(void* const* d_in, const int* in_sizes, int n_in,
                              void* d_out, int out_size, void* d_ws, size_t ws_size,
                              hipStream_t stream) {
  const float* x     = (const float*)d_in[0];
  const float* W_in  = (const float*)d_in[1];
  const float* b_in  = (const float*)d_in[2];
  const float* W_res = (const float*)d_in[3];
  const float* b_res = (const float*)d_in[4];
  const float* W_out = (const float*)d_in[5];
  const float* b_out = (const float*)d_in[6];
  float* out = (float*)d_out;

  // workspace layout (bf16, 16B-aligned): WinT | WresT(4 layers) | WoutT
  char* ws = (char*)d_ws;
  __hip_bfloat16* WinT  = (__hip_bfloat16*)ws;                            // 512*32
  __hip_bfloat16* WresT = (__hip_bfloat16*)(ws + 512 * 32 * 2);           // 4*512*512
  __hip_bfloat16* WoutT = (__hip_bfloat16*)(ws + 512 * 32 * 2
                                               + (size_t)4 * 512 * 512 * 2); // 1504*512

  cvt_transpose<<<(32 * 512 + 255) / 256, 256, 0, stream>>>(W_in, WinT, 32, 512);
  for (int l = 0; l < 4; ++l) {
    cvt_transpose<<<(512 * 512 + 255) / 256, 256, 0, stream>>>(
        W_res + (size_t)l * 512 * 512, WresT + (size_t)l * 512 * 512, 512, 512);
  }
  cvt_transpose<<<(512 * 1504 + 255) / 256, 256, 0, stream>>>(W_out, WoutT, 512, 1504);

  const size_t smem = (size_t)BM * HPAD * sizeof(__hip_bfloat16)    // 66,560 B
                    + (size_t)BM * TPAD * sizeof(float);            // 49,152 B
  hipFuncSetAttribute((const void*)nsf_fused,
                      hipFuncAttributeMaxDynamicSharedMemorySize, (int)smem);
  nsf_fused<<<BATCH_ / BM, 256, smem, stream>>>(x, b_in, b_res, b_out,
                                                WinT, WresT, WoutT, out);
}
